// Sequence_57260503990382
// MI455X (gfx1250) — matrix operand order repaired and checked by reading, not compiled, not run on hardware
//
#include <hip/hip_runtime.h>
#include <hip/hip_bf16.h>
#include <math.h>

// Problem constants (from reference)
#define BB   16      // batch
#define TT   32      // time steps
#define HH   512     // hidden channels
#define HW   32      // spatial
#define NP   (HW*HW)         // 1024 pixels
#define NTOT (BB*NP)         // 16384 = GEMM M (pixels)
#define MG   (4*HH)          // 2048 gate rows = GEMM N
#define KC   513             // 1 (x) + 512 (h)

typedef __attribute__((ext_vector_type(16))) _Float16 v16h;
typedef __attribute__((ext_vector_type(8)))  _Float16 v8h;
typedef __attribute__((ext_vector_type(8)))  float    v8f;

union Frag { v16h v; v8h h[2]; };

__device__ __forceinline__ float sigmoidf_(float x) { return 1.0f / (1.0f + __expf(-x)); }

// -------- CDNA5 async global->LDS path (guarded; sync fallback otherwise) ----
#if __has_builtin(__builtin_amdgcn_global_load_async_to_lds_b128)
#define HAS_ASYNC_LDS 1
typedef int v4i __attribute__((vector_size(16)));
typedef __attribute__((address_space(1))) v4i* gv4i_p;   // global AS pointer
typedef __attribute__((address_space(3))) v4i* lv4i_p;   // LDS AS pointer
__device__ __forceinline__ void async_cp16(const void* g, void* l) {
    __builtin_amdgcn_global_load_async_to_lds_b128((gv4i_p)(void*)g, (lv4i_p)l, 0, 0);
}
__device__ __forceinline__ void wait_async0() {
#if __has_builtin(__builtin_amdgcn_s_wait_asynccnt)
    __builtin_amdgcn_s_wait_asynccnt(0);
#else
    asm volatile("s_wait_asynccnt 0x0" ::: "memory");
#endif
}
#else
#define HAS_ASYNC_LDS 0
#endif

// ---------------------------------------------------------------------------
// Prep: Wc[o, 1+c] -> f16 B-matrix Wh[o][c] (K-contiguous rows), o in [0,2048)
// ---------------------------------------------------------------------------
__global__ void prep_weights(const float* __restrict__ Wc, _Float16* __restrict__ Wh) {
    int i = blockIdx.x * blockDim.x + threadIdx.x;   // 2048*512 = 1M elements
    if (i < MG * HH) {
        int row = i >> 9;          // /512
        int c   = i & 511;
        Wh[i] = (_Float16)Wc[row * KC + 1 + c];
    }
}

// Zero h0 (f16 n-major state, read at t=0) and c (f32 n-major state)
__global__ void init_state(unsigned* __restrict__ h0u, unsigned* __restrict__ cu) {
    int i = blockIdx.x * blockDim.x + threadIdx.x;
    const int NHU = (HH * NTOT) / 2;   // h0: 16MB f16 = 4M uints
    const int NCU = HH * NTOT;         // c : 32MB f32 = 8M uints
    if (i < NHU)            h0u[i] = 0u;
    else if (i < NHU + NCU) cu[i - NHU] = 0u;
}

// ---------------------------------------------------------------------------
// Fused gate-GEMM + LSTM pointwise, one timestep.
//   gatesT[n, o] = sum_c h_nm[n][c] * Wh[o][c]   (WMMA: A=h tile, B=W tile)
//                + Wc[o][0]*x_t[n] + b_conv[o]
// Block: 256 threads = 8 waves. Tile: M=128 pixels x N=64 (16 ch x 4 gates).
// Wave w owns pixel rows [w*16, w*16+16) and all 4 gate accumulators; lane
// maps to channel, VGPR index to pixel -> LSTM pointwise is fully in-lane.
// Both LDS tiles are straight row-major copies -> async-to-LDS + double buffer.
// h ping-pongs across timesteps (no intra-step RAW across blocks).
// ---------------------------------------------------------------------------
__global__ void __launch_bounds__(256)
lstm_step(const float* __restrict__ x,        // (B,T,HW,HW)
          const float* __restrict__ Wc,       // (2048,513): col 0 used here
          const float* __restrict__ bconv,    // (2048)
          const _Float16* __restrict__ Wh,    // (2048,512) f16 (B operand)
          const _Float16* __restrict__ hin,   // (16384,512) f16 n-major
          _Float16* __restrict__ hout,        // (16384,512) f16 n-major
          float* __restrict__ cst,            // (16384,512) f32 n-major
          int t)
{
    __shared__ __align__(16) _Float16 As[2][128 * 32];   // pixel tiles, 2 x 8KB
    __shared__ __align__(16) _Float16 Bs[2][64 * 32];    // weight tiles, 2 x 4KB

    const int tid  = threadIdx.x;
    const int lane = tid & 31;
    const int w    = tid >> 5;            // wave 0..7
    const int hi   = lane >> 4;           // half-wave (K/M group select)
    const int nb0  = blockIdx.x * 128;    // pixel-tile base (M)
    const int ch0  = blockIdx.y * 16;     // channel-group base (within a gate)

    // Per-thread copy mapping (constant across K loop)
    // A tile: 128 rows x 32 halves = 512 b128 chunks -> 2 per thread
    const int arow0 = tid >> 2;               // chunk 0 row (0..63)
    const int aseg  = tid & 3;                // 8-half segment 0..3
    // B tile: 64 rows x 32 halves = 256 b128 chunks -> 1 per thread
    const int brow  = tid >> 2;               // 0..63
    const long orow = (long)(brow >> 4) * HH + ch0 + (brow & 15); // gate-strided

    v8f acc[4];
    #pragma unroll
    for (int g = 0; g < 4; ++g) acc[g] = v8f{};

#if HAS_ASYNC_LDS
    // ---- prologue: async-copy K-tile 0 into buffer 0
    {
        async_cp16(&hin[(size_t)(nb0 + arow0) * HH + aseg * 8],       &As[0][arow0 * 32 + aseg * 8]);
        async_cp16(&hin[(size_t)(nb0 + arow0 + 64) * HH + aseg * 8],  &As[0][(arow0 + 64) * 32 + aseg * 8]);
        async_cp16(&Wh[orow * HH + aseg * 8],                         &Bs[0][brow * 32 + aseg * 8]);
    }
    int buf = 0;
#endif

    for (int kk = 0; kk < HH; kk += 32) {
#if HAS_ASYNC_LDS
        wait_async0();        // this wave's tile-copies done
        __syncthreads();      // everyone's copies visible; prev reads of buf^1 done

        if (kk + 32 < HH) {   // issue next tile into the other buffer
            const int kn = kk + 32;
            async_cp16(&hin[(size_t)(nb0 + arow0) * HH + kn + aseg * 8],      &As[buf ^ 1][arow0 * 32 + aseg * 8]);
            async_cp16(&hin[(size_t)(nb0 + arow0 + 64) * HH + kn + aseg * 8], &As[buf ^ 1][(arow0 + 64) * 32 + aseg * 8]);
            async_cp16(&Wh[orow * HH + kn + aseg * 8],                        &Bs[buf ^ 1][brow * 32 + aseg * 8]);
        }
        const _Float16* asrc = As[buf];
        const _Float16* bsrc = Bs[buf];
#else
        __syncthreads();      // previous iteration done reading LDS
        *(v8h*)&As[0][arow0 * 32 + aseg * 8] =
            *(const v8h*)&hin[(size_t)(nb0 + arow0) * HH + kk + aseg * 8];
        *(v8h*)&As[0][(arow0 + 64) * 32 + aseg * 8] =
            *(const v8h*)&hin[(size_t)(nb0 + arow0 + 64) * HH + kk + aseg * 8];
        *(v8h*)&Bs[0][brow * 32 + aseg * 8] =
            *(const v8h*)&Wh[orow * HH + kk + aseg * 8];
        __syncthreads();
        const _Float16* asrc = As[0];
        const _Float16* bsrc = Bs[0];
#endif
        // L2 prefetch two tiles ahead (global_prefetch_b8)
        if (kk + 64 < HH)
            __builtin_prefetch(&hin[(size_t)(nb0 + arow0) * HH + kk + 64 + aseg * 8], 0, 1);

        // ---- A fragment (pixels): lane row m = lane%16 of wave's 16-row strip.
        // f16 A layout: halves[0..8) = K[hi*8..), halves[8..16) = K[16+hi*8..)
        Frag af;
        {
            const int nl = w * 16 + (lane & 15);
            af.h[0] = *(const v8h*)&asrc[nl * 32 + hi * 8];
            af.h[1] = *(const v8h*)&asrc[nl * 32 + 16 + hi * 8];
        }

        // ---- 4 gate WMMAs sharing the A fragment.
        // f16 B layout: lane = column, halves[0..8) = K[hi*8..), halves[8..16) = K[16+hi*8..)
        const int j = lane & 15;
        #pragma unroll
        for (int g = 0; g < 4; ++g) {
            Frag bf;
            bf.h[0] = *(const v8h*)&bsrc[g * 512 + j * 32 + hi * 8];
            bf.h[1] = *(const v8h*)&bsrc[g * 512 + j * 32 + 16 + hi * 8];
            acc[g] = __builtin_amdgcn_wmma_f32_16x16x32_f16(
                false, af.v, false, bf.v, (short)0, acc[g], false, false);
        }
#if HAS_ASYNC_LDS
        buf ^= 1;
#endif
    }

    // ---- Epilogue: rank-1 x-term + bias, LSTM pointwise, state update.
    // D layout: VGPR r -> pixel M = r + hi*8; N = lane%16 -> channel.
    const int ch     = ch0 + (lane & 15);
    const int base_n = nb0 + w * 16 + hi * 8;

    const float bi = bconv[ch],          wi = Wc[(size_t)ch * KC];
    const float bff= bconv[HH + ch],     wf = Wc[(size_t)(HH + ch) * KC];
    const float bo = bconv[2*HH + ch],   wo = Wc[(size_t)(2*HH + ch) * KC];
    const float bg = bconv[3*HH + ch],   wg = Wc[(size_t)(3*HH + ch) * KC];

    #pragma unroll
    for (int r = 0; r < 8; ++r) {
        const int n = base_n + r;
        const int b = n >> 10;
        const int p = n & 1023;
        const float xv = x[(size_t)b * (TT * NP) + (size_t)t * NP + p];

        float iv = acc[0][r] + bi  + wi * xv;
        float fv = acc[1][r] + bff + wf * xv;
        float ov = acc[2][r] + bo  + wo * xv;
        float gv = acc[3][r] + bg  + wg * xv;

        const size_t sidx = (size_t)n * HH + ch;   // n-major state
        float co = cst[sidx];
        float cn = sigmoidf_(fv) * co + sigmoidf_(iv) + tanhf(gv);
        float hn = sigmoidf_(ov) + tanhf(cn);
        cst[sidx]  = cn;
        hout[sidx] = (_Float16)hn;
    }
}

// ---------------------------------------------------------------------------
// 3x3 SAME conv (H -> 1): with n-major h, each (pixel, tap) is a contiguous
// 512-long dot product. One thread per output pixel, weights transposed in LDS.
// ---------------------------------------------------------------------------
__global__ void __launch_bounds__(256)
post_conv(const _Float16* __restrict__ h,     // (16384, 512) n-major
          const float* __restrict__ Wpost,    // (512,3,3)
          const float* __restrict__ bpost,    // (1,)
          float* __restrict__ out,            // (16,32,32,32)
          int t)
{
    __shared__ float wl[9 * HH];    // wl[tap][c] = Wpost[c][tap], 18KB

    const int tid = threadIdx.x;
    for (int i = tid; i < 9 * HH; i += 256) {
        int tap = i >> 9;      // /512
        int c   = i & 511;
        wl[i] = Wpost[c * 9 + tap];
    }
    __syncthreads();

    const int id = blockIdx.x * 256 + tid;   // 0..16383 pixel id
    const int b  = id >> 10;
    const int p  = id & 1023;
    const int y  = p >> 5;
    const int x0 = p & 31;

    float acc = 0.f;
    #pragma unroll
    for (int ky = 0; ky < 3; ++ky) {
        const int yy = y + ky - 1;
        if (yy < 0 || yy >= HW) continue;
        #pragma unroll
        for (int kx = 0; kx < 3; ++kx) {
            const int xx = x0 + kx - 1;
            if (xx < 0 || xx >= HW) continue;
            const _Float16* hp = &h[(size_t)(b * NP + yy * 32 + xx) * HH];
            const float*    wp = &wl[(ky * 3 + kx) * HH];
            float s = 0.f;
            for (int c = 0; c < HH; c += 8) {
                v8h hv = *(const v8h*)&hp[c];
                #pragma unroll
                for (int q = 0; q < 8; ++q) s += (float)hv[q] * wp[c + q];
            }
            acc += s;
        }
    }
    out[(size_t)b * (TT * NP) + (size_t)t * NP + p] = acc + bpost[0];
}

// ---------------------------------------------------------------------------
extern "C" void kernel_launch(void* const* d_in, const int* in_sizes, int n_in,
                              void* d_out, int out_size, void* d_ws, size_t ws_size,
                              hipStream_t stream) {
    const float* x      = (const float*)d_in[0];   // (16,32,32,32)
    const float* Wconv  = (const float*)d_in[1];   // (2048,513,1,1)
    const float* bconv  = (const float*)d_in[2];   // (2048,)
    const float* Wpost  = (const float*)d_in[3];   // (1,512,3,3)
    const float* bpost  = (const float*)d_in[4];   // (1,)
    float* out          = (float*)d_out;           // (16,32,32,32)

    // Workspace layout
    char* ws = (char*)d_ws;
    const size_t WH_BYTES = (size_t)MG * HH * sizeof(_Float16);   //  2 MB
    const size_t H_BYTES  = (size_t)HH * NTOT * sizeof(_Float16); // 16 MB
    _Float16* Wh = (_Float16*)ws;
    _Float16* h0 = (_Float16*)(ws + WH_BYTES);
    _Float16* h1 = (_Float16*)(ws + WH_BYTES + H_BYTES);
    float*    cs = (float*)   (ws + WH_BYTES + 2 * H_BYTES);      // 32 MB

    prep_weights<<<(MG * HH + 255) / 256, 256, 0, stream>>>(Wconv, Wh);
    {
        const int n_uints = (HH * NTOT) / 2 + HH * NTOT;
        init_state<<<(n_uints + 255) / 256, 256, 0, stream>>>((unsigned*)h0, (unsigned*)cs);
    }

    dim3 gGemm(NTOT / 128, HH / 16);   // (128, 32)

    for (int t = 0; t < TT; ++t) {
        _Float16* hin  = (t & 1) ? h1 : h0;
        _Float16* hout = (t & 1) ? h0 : h1;
        lstm_step<<<gGemm, 256, 0, stream>>>(x, Wconv, bconv, Wh, hin, hout, cs, t);
        post_conv<<<NTOT / 256, 256, 0, stream>>>(hout, Wpost, bpost, out, t);
    }
}
